// QuantLinear_72060961292367
// MI455X (gfx1250) — 8-bit matrix operand order repaired and checked by reading, compiled, not run on hardware
//
#include <hip/hip_runtime.h>
#include <hip/hip_bf16.h>


typedef __attribute__((ext_vector_type(16))) int   v16i;
typedef __attribute__((ext_vector_type(8)))  float v8f;
typedef int v4i_vs __attribute__((vector_size(16)));

constexpr int M  = 8192;
constexpr int K  = 4096;
constexpr int N  = 11008;
constexpr int KB = K / 128;
constexpr int NB = N / 128;
constexpr int MBt = M / 128;
constexpr int LDA = 144;

#if defined(__has_builtin)
#  if __has_builtin(__builtin_amdgcn_global_load_async_to_lds_b128)
#    define USE_ASYNC_COPY 1
#  endif
#endif
#ifndef USE_ASYNC_COPY
#  define USE_ASYNC_COPY 0
#endif

#if USE_ASYNC_COPY
__device__ __forceinline__ void async_copy16(void* lds_dst, const void* gsrc) {
  typedef __attribute__((address_space(1))) v4i_vs GT;
  typedef __attribute__((address_space(3))) v4i_vs LT;
  __builtin_amdgcn_global_load_async_to_lds_b128(
      (GT*)(uintptr_t)gsrc,
      (LT*)(unsigned)(uintptr_t)lds_dst, 0, 0);
}
template <int Nw>
__device__ __forceinline__ void wait_async() {
#  if __has_builtin(__builtin_amdgcn_s_wait_asynccnt)
  __builtin_amdgcn_s_wait_asynccnt(Nw);
#  else
  asm volatile("s_wait_asynccnt %0" ::"i"(Nw) : "memory");
#  endif
}
#endif

__device__ __forceinline__ unsigned char f32_to_e4m3(float v) {
  unsigned u = __float_as_uint(v);
  unsigned s = (u >> 24) & 0x80u;
  float av = fabsf(v);
  if (!(av > 0.0f)) return (unsigned char)s;
  if (av > 448.0f) av = 448.0f;
  unsigned ua = __float_as_uint(av);
  int fexp = (int)(ua >> 23) - 127;
  int be   = fexp < -6 ? -6 : fexp;
  float scaled = av * __uint_as_float((unsigned)(127 + 3 - be) << 23);
  int r = (int)rintf(scaled);
  if (r >= 16) { be += 1; r = 8; }
  unsigned code = (r >= 8) ? (((unsigned)(be + 7) << 3) | (unsigned)(r - 8))
                           : (unsigned)r;
  return (unsigned char)(code | s);
}

__global__ void quant_x_kernel(const float* __restrict__ x,
                               unsigned char* __restrict__ xq,
                               float* __restrict__ xs) {
  const int lane = threadIdx.x & 31;
  const int w    = threadIdx.x >> 5;
  const long long blk = (long long)blockIdx.x * 8 + w;
  const int m  = (int)(blk / KB);
  const int kb = (int)(blk % KB);
  const size_t base = (size_t)m * K + (size_t)kb * 128;

  float4 v = *(const float4*)(x + base + lane * 4);
  v.x = __bfloat162float(__float2bfloat16(v.x));
  v.y = __bfloat162float(__float2bfloat16(v.y));
  v.z = __bfloat162float(__float2bfloat16(v.z));
  v.w = __bfloat162float(__float2bfloat16(v.w));

  float amax = fmaxf(fmaxf(fabsf(v.x), fabsf(v.y)),
                     fmaxf(fabsf(v.z), fabsf(v.w)));
#pragma unroll
  for (int off = 16; off >= 1; off >>= 1)
    amax = fmaxf(amax, __shfl_xor(amax, off, 32));

  float r = fmaxf(amax, 1e-30f) / 448.0f;
  unsigned ub = __float_as_uint(r);
  int ex = (int)(ub >> 23) - 127;
  if (ub & 0x7FFFFFu) ++ex;
  ex = ex < -126 ? -126 : (ex > 126 ? 126 : ex);
  const float scale = __uint_as_float((unsigned)(ex + 127) << 23);
  const float inv   = __uint_as_float((unsigned)(127 - ex) << 23);

  if (lane == 0) xs[(size_t)m * KB + kb] = scale;

  unsigned q =  (unsigned)f32_to_e4m3(v.x * inv)
             | ((unsigned)f32_to_e4m3(v.y * inv) << 8)
             | ((unsigned)f32_to_e4m3(v.z * inv) << 16)
             | ((unsigned)f32_to_e4m3(v.w * inv) << 24);
  *(unsigned*)(xq + base + lane * 4) = q;
}

__global__ void quant_w_kernel(const float* __restrict__ w,
                               unsigned char* __restrict__ wq) {
  size_t i = ((size_t)blockIdx.x * 256 + threadIdx.x) * 4;
  float4 v = *(const float4*)(w + i);
  unsigned q =  (unsigned)f32_to_e4m3(v.x)
             | ((unsigned)f32_to_e4m3(v.y) << 8)
             | ((unsigned)f32_to_e4m3(v.z) << 16)
             | ((unsigned)f32_to_e4m3(v.w) << 24);
  *(unsigned*)(wq + i) = q;
}

__global__ void __launch_bounds__(256)
gemm_kernel(const unsigned char* __restrict__ xq,
            const unsigned char* __restrict__ wq,
            const float* __restrict__ xs,
            const float* __restrict__ wsc,
            __hip_bfloat16* __restrict__ out) {
#if USE_ASYNC_COPY
  __shared__ __align__(16) unsigned char As[2][128 * LDA];
  __shared__ __align__(16) unsigned char Bs[2][128 * LDA];
  __shared__ __align__(16) float Xs[2][128];
#else
  __shared__ __align__(16) unsigned char As1[128 * LDA];
  __shared__ __align__(16) unsigned char Bs1[128 * LDA];
  __shared__ __align__(16) float Xs1[128];
#endif

  const int nb   = blockIdx.x;
  const int mb   = blockIdx.y;
  const int t    = threadIdx.x;
  const int lane = t & 31;
  const int wid  = t >> 5;
  const int g    = lane >> 4;
  const int lr   = lane & 15;
  const int wm   = wid & 3;
  const int wn   = wid >> 2;

  v8f acc[2][4];
  v8f zero = {};
#pragma unroll
  for (int i = 0; i < 2; ++i)
#pragma unroll
    for (int j = 0; j < 4; ++j) acc[i][j] = zero;

  auto compute_step = [&](const unsigned char* Ab, const unsigned char* Bb,
                          const float* Xb, float sw) {
    v16i afrag[2];
#pragma unroll
    for (int mt = 0; mt < 2; ++mt) {
      const unsigned char* ap = &Ab[(wm * 32 + mt * 16 + lr) * LDA];
#pragma unroll
      for (int v = 0; v < 16; ++v) {
        int half = v >> 3, v8 = v & 7, pair = v8 >> 1, sub = v8 & 1;
        int k = (v >> 2) * 32 + g * 16 + (v & 3) * 4;
        afrag[mt][v] = *(const int*)(ap + k);
      }
    }
    float sxw[2][8];
#pragma unroll
    for (int mt = 0; mt < 2; ++mt)
#pragma unroll
      for (int r = 0; r < 8; ++r)
        sxw[mt][r] = Xb[wm * 32 + mt * 16 + g * 8 + r] * sw;

#pragma unroll
    for (int nt = 0; nt < 4; ++nt) {
      v16i bfrag;
      const unsigned char* bp = &Bb[(wn * 64 + nt * 16 + lr) * LDA];
#pragma unroll
      for (int v = 0; v < 16; ++v) {
        int q = v >> 2, vq = v & 3;
        int k = q * 32 + g * 16 + vq * 4;
        bfrag[v] = *(const int*)(bp + k);
      }
#pragma unroll
      for (int mt = 0; mt < 2; ++mt) {
        v8f part = __builtin_amdgcn_wmma_f32_16x16x128_fp8_fp8(
            afrag[mt], bfrag, (short)0, zero, false, false);
#pragma unroll
        for (int r = 0; r < 8; ++r)
          acc[mt][nt][r] = fmaf(part[r], sxw[mt][r], acc[mt][nt][r]);
      }
    }
  };

#if USE_ASYNC_COPY
  auto prefetch = [&](int buf, int kb) {
#pragma unroll
    for (int it = 0; it < 4; ++it) {
      int idx = t + it * 256;
      int row = idx >> 3;
      int ch  = (idx & 7) << 4;
      async_copy16(&As[buf][row * LDA + ch],
                   xq + (size_t)(mb * 128 + row) * K + (size_t)kb * 128 + ch);
      async_copy16(&Bs[buf][row * LDA + ch],
                   wq + (size_t)(nb * 128 + row) * K + (size_t)kb * 128 + ch);
    }
  };

  prefetch(0, 0);
  for (int kb = 0; kb < KB; ++kb) {
    const int cur = kb & 1;
    if (kb + 1 < KB) prefetch(cur ^ 1, kb + 1);
    if (t < 128) Xs[cur][t] = xs[(size_t)(mb * 128 + t) * KB + kb];
    if (kb + 1 < KB) wait_async<8>();
    else             wait_async<0>();
    __syncthreads();
    const float sw = wsc[(size_t)nb * KB + kb];
    compute_step(As[cur], Bs[cur], Xs[cur], sw);
    __syncthreads();
  }
#else
  for (int kb = 0; kb < KB; ++kb) {
#pragma unroll
    for (int it = 0; it < 4; ++it) {
      int idx = t + it * 256;
      int row = idx >> 3;
      int ch  = (idx & 7) << 4;
      *(uint4*)&As1[row * LDA + ch] =
          *(const uint4*)(xq + (size_t)(mb * 128 + row) * K + (size_t)kb * 128 + ch);
      *(uint4*)&Bs1[row * LDA + ch] =
          *(const uint4*)(wq + (size_t)(nb * 128 + row) * K + (size_t)kb * 128 + ch);
    }
    if (t < 128) Xs1[t] = xs[(size_t)(mb * 128 + t) * KB + kb];
    __syncthreads();
    const float sw = wsc[(size_t)nb * KB + kb];
    compute_step(As1, Bs1, Xs1, sw);
    __syncthreads();
  }
#endif

#pragma unroll
  for (int mt = 0; mt < 2; ++mt)
#pragma unroll
    for (int nt = 0; nt < 4; ++nt) {
      int col = nb * 128 + wn * 64 + nt * 16 + lr;
#pragma unroll
      for (int r = 0; r < 8; ++r) {
        int row = mb * 128 + wm * 32 + mt * 16 + g * 8 + r;
        out[(size_t)row * N + col] = __float2bfloat16(acc[mt][nt][r]);
      }
    }
}

extern "C" void kernel_launch(void* const* d_in, const int* in_sizes, int n_in,
                              void* d_out, int out_size, void* d_ws, size_t ws_size,
                              hipStream_t stream) {
  const float* x    = (const float*)d_in[0];
  const float* w_q  = (const float*)d_in[1];
  const float* wsc  = (const float*)d_in[2];
  __hip_bfloat16* out = (__hip_bfloat16*)d_out;

  unsigned char* ws = (unsigned char*)d_ws;
  size_t off = 0;
  unsigned char* xq = ws + off; off += (size_t)M * K;
  unsigned char* wq = ws + off; off += (size_t)N * K;
  off = (off + 255) & ~(size_t)255;
  float* xsc = (float*)(ws + off);

  quant_x_kernel<<<(M * KB) / 8, 256, 0, stream>>>(x, xq, xsc);
  quant_w_kernel<<<(N * K) / 1024, 256, 0, stream>>>(w_q, wq);

  dim3 grid(NB, MBt);
  gemm_kernel<<<grid, 256, 0, stream>>>(xq, wq, xsc, wsc, out);
}
